// GATModel_72619307041224
// MI455X (gfx1250) — matrix operand order repaired and checked by reading, not compiled, not run on hardware
//
#include <hip/hip_runtime.h>

typedef __attribute__((ext_vector_type(16))) _Float16 v16h;
typedef __attribute__((ext_vector_type(8)))  float    v8f;
typedef __attribute__((ext_vector_type(4)))  float    v4f;

#define N_NODES 100000
#define N_EDGES 1000000
#define D_IN    128
#define D_OUT   64
#define NEG_SLOPE 0.2f

// ---- monotonic float<->u32 key for atomicMax-based segment max ----
__device__ __forceinline__ unsigned fkey(float f) {
    int b = __float_as_int(f);
    unsigned u = (unsigned)b;
    return (b < 0) ? ~u : (u | 0x80000000u);
}
__device__ __forceinline__ float fdec(unsigned k) {
    unsigned b = (k & 0x80000000u) ? (k & 0x7FFFFFFFu) : ~k;
    return __int_as_float((int)b);
}

// ---- init: zero output accumulator + denom, set emax keys to -1e30 ----
__global__ void k_init(float* __restrict__ out, unsigned* __restrict__ keymax,
                       float* __restrict__ denom) {
    int i = blockIdx.x * blockDim.x + threadIdx.x;
    if (i < N_NODES * D_OUT) out[i] = 0.0f;
    if (i < N_NODES) { keymax[i] = fkey(-1e30f); denom[i] = 0.0f; }
}

// ---- projection GEMM h = x @ W^T via WMMA f16->f32, fused el/er ----
// block = 256 threads = 8 waves; wave handles 16 rows x 64 cols.
__global__ __launch_bounds__(256) void k_gemm(
        const float* __restrict__ x, const float* __restrict__ W,
        const float* __restrict__ attn_l, const float* __restrict__ attn_r,
        float* __restrict__ h, float* __restrict__ el, float* __restrict__ er) {
    __shared__ __align__(32) _Float16 Wh[D_OUT * D_IN];   // 16 KB of the 320 KB WGP LDS
    for (int i = threadIdx.x; i < D_OUT * D_IN; i += 256)
        Wh[i] = (_Float16)W[i];
    __syncthreads();

    const int wave = threadIdx.x >> 5;
    const int lane = threadIdx.x & 31;
    const int m    = lane & 15;       // N column (B/C) or M row (A)
    const int hi   = lane >> 4;       // half-wave selector
    const int rowBase = (blockIdx.x * 8 + wave) * 16;
    const int row  = rowBase + m;
    const int rrow = row < N_NODES ? row : (N_NODES - 1);  // clamp so EXEC stays full
    const float* xr = x + (size_t)rrow * D_IN;

    v8f acc[4] = {};
    #pragma unroll
    for (int kk = 0; kk < 4; ++kk) {          // K chunks of 32
        // A fragment: elements 0..15 -> K = kk*32 + hi*16 + e (contiguous)
        const int ka = kk * 32 + hi * 16;
        v4f a0 = *(const v4f*)(xr + ka);
        v4f a1 = *(const v4f*)(xr + ka + 4);
        v4f a2 = *(const v4f*)(xr + ka + 8);
        v4f a3 = *(const v4f*)(xr + ka + 12);
        v16h a;
        #pragma unroll
        for (int e = 0; e < 4; ++e) {
            a[e]      = (_Float16)a0[e];
            a[4 + e]  = (_Float16)a1[e];
            a[8 + e]  = (_Float16)a2[e];
            a[12 + e] = (_Float16)a3[e];
        }
        // B fragments from LDS: lane holds column n, K = kk*32 + hi*16 .. +15 contiguous
        const int kb = kk * 32 + hi * 16;
        #pragma unroll
        for (int nt = 0; nt < 4; ++nt) {
            const int n = nt * 16 + m;
            v16h b = *(const v16h*)&Wh[n * D_IN + kb];
            acc[nt] = __builtin_amdgcn_wmma_f32_16x16x32_f16(
                false, a, false, b, (short)0, acc[nt], false, false);
        }
    }

    // store h: VGPR r holds M = rowBase + hi*8 + r, column = nt*16 + m
    #pragma unroll
    for (int r = 0; r < 8; ++r) {
        const int mr = rowBase + hi * 8 + r;
        if (mr < N_NODES) {
            #pragma unroll
            for (int nt = 0; nt < 4; ++nt)
                h[(size_t)mr * D_OUT + nt * 16 + m] = acc[nt][r];
        }
    }

    // fused el/er: reduce over the 16 N-lanes of each half-wave
    #pragma unroll
    for (int r = 0; r < 8; ++r) {
        float pl = 0.0f, pr = 0.0f;
        #pragma unroll
        for (int nt = 0; nt < 4; ++nt) {
            pl += acc[nt][r] * attn_l[nt * 16 + m];
            pr += acc[nt][r] * attn_r[nt * 16 + m];
        }
        #pragma unroll
        for (int off = 8; off >= 1; off >>= 1) {
            pl += __shfl_xor(pl, off, 16);
            pr += __shfl_xor(pr, off, 16);
        }
        const int mr = rowBase + hi * 8 + r;
        if (m == 0 && mr < N_NODES) { el[mr] = pl; er[mr] = pr; }
    }
}

// ---- edge pass 1: leaky-relu logit + segment max via u32-key atomicMax ----
__global__ void k_edge1(const int* __restrict__ src, const int* __restrict__ dst,
                        const float* __restrict__ el, const float* __restrict__ er,
                        float* __restrict__ e, unsigned* __restrict__ keymax) {
    int i = blockIdx.x * blockDim.x + threadIdx.x;
    if (i >= N_EDGES) return;
    float v = el[src[i]] + er[dst[i]];
    v = v > 0.0f ? v : NEG_SLOPE * v;
    e[i] = v;
    atomicMax(&keymax[dst[i]], fkey(v));
}

// ---- edge pass 2: scatter exp(e - emax[dst]) * h[src] ; accumulate denom ----
// 64 threads per edge: coalesced h gather + coalesced f32 atomics (L2-resident out).
__global__ __launch_bounds__(256) void k_edge2(
        const int* __restrict__ src, const int* __restrict__ dst,
        const float* __restrict__ e, const unsigned* __restrict__ keymax,
        const float* __restrict__ h, float* __restrict__ denom,
        float* __restrict__ out) {
    long long t = (long long)blockIdx.x * blockDim.x + threadIdx.x;
    int i = (int)(t >> 6);
    int d = (int)(t & 63);
    if (i >= N_EDGES) return;
    int s = src[i], q = dst[i];
    float ex = __expf(e[i] - fdec(keymax[q]));
    if (d == 0) atomicAdd(&denom[q], ex);
    atomicAdd(&out[(size_t)q * D_OUT + d], ex * h[(size_t)s * D_OUT + d]);
}

// ---- finalize: normalize by segment sum, add bias ----
__global__ void k_final(float* __restrict__ out, const float* __restrict__ denom,
                        const float* __restrict__ bias) {
    int i = blockIdx.x * blockDim.x + threadIdx.x;
    if (i >= N_NODES * D_OUT) return;
    int n = i >> 6, d = i & 63;
    float dn = denom[n];
    out[i] = out[i] / (dn > 1e-9f ? dn : 1e-9f) + bias[d];
}

extern "C" void kernel_launch(void* const* d_in, const int* in_sizes, int n_in,
                              void* d_out, int out_size, void* d_ws, size_t ws_size,
                              hipStream_t stream) {
    const float* x      = (const float*)d_in[0];
    const float* W      = (const float*)d_in[1];
    const float* attn_l = (const float*)d_in[2];
    const float* attn_r = (const float*)d_in[3];
    const float* bias   = (const float*)d_in[4];
    const int*   src    = (const int*)d_in[5];
    const int*   dst    = (const int*)d_in[6];
    float* out = (float*)d_out;

    // workspace layout (~31.2 MB)
    float*    h      = (float*)d_ws;
    float*    el     = h + (size_t)N_NODES * D_OUT;
    float*    er     = el + N_NODES;
    float*    e      = er + N_NODES;
    unsigned* keymax = (unsigned*)(e + N_EDGES);
    float*    denom  = (float*)(keymax + N_NODES);

    const int B = 256;
    k_init <<<(N_NODES * D_OUT + B - 1) / B, B, 0, stream>>>(out, keymax, denom);
    k_gemm <<<(N_NODES + 127) / 128, B, 0, stream>>>(x, W, attn_l, attn_r, h, el, er);
    k_edge1<<<(N_EDGES + B - 1) / B, B, 0, stream>>>(src, dst, el, er, e, keymax);
    long long t2 = (long long)N_EDGES * 64;
    k_edge2<<<(unsigned)((t2 + B - 1) / B), B, 0, stream>>>(src, dst, e, keymax, h, denom, out);
    k_final<<<(N_NODES * D_OUT + B - 1) / B, B, 0, stream>>>(out, denom, bias);
}
